// Attention2D_87402584474004
// MI455X (gfx1250) — matrix operand order repaired and checked by reading, not compiled, not run on hardware
//
#include <hip/hip_runtime.h>

// ---------------------------------------------------------------------------
// Problem constants (B=8, L=1024, D=512, H=8, d=64)
// ---------------------------------------------------------------------------
#define B_SZ 8
#define L_SZ 1024
#define D_MODEL 512
#define H_SZ 8
#define D_HEAD 64

typedef __attribute__((ext_vector_type(16))) __bf16        v16bf;
typedef __attribute__((ext_vector_type(8)))  float         v8f;
typedef __attribute__((ext_vector_type(8)))  unsigned int  v8u;

// ---- float -> bf16 (round-to-nearest-even-ish) helpers --------------------
__device__ __forceinline__ unsigned int f2bf_pack(float lo, float hi) {
  union { float f; unsigned int u; } a, b;
  a.f = lo; b.f = hi;
  unsigned int ua = (a.u + 0x7FFFu + ((a.u >> 16) & 1u)) >> 16;
  unsigned int ub = (b.u + 0x7FFFu + ((b.u >> 16) & 1u)) >> 16;
  return (ua & 0xFFFFu) | (ub << 16);
}
__device__ __forceinline__ unsigned short f2bf(float x) {
  union { float f; unsigned int u; } a; a.f = x;
  return (unsigned short)((a.u + 0x7FFFu + ((a.u >> 16) & 1u)) >> 16);
}

// LDS byte offset of a __shared__ object: flat shared pointers are
// {aperture_hi32, lds_offset32}, so the low 32 bits are the LDS address.
__device__ __forceinline__ unsigned lds_off(const void* p) {
  return (unsigned)(size_t)p;
}

// ---- one WMMA issue --------------------------------------------------------
__device__ __forceinline__ v8f wmma_bf16(const v8u& a, const v8u& b, v8f c) {
  return __builtin_amdgcn_wmma_f32_16x16x32_bf16(
      /*neg_a=*/false, __builtin_bit_cast(v16bf, a),
      /*neg_b=*/false, __builtin_bit_cast(v16bf, b),
      /*c_mod=*/(short)0, c, /*reuse_a=*/false, /*reuse_b=*/false);
}

// Per-wave 16(M)x64(N) x 32(K) MMA step from LDS tiles.
// sA: [128][16] uint = bf16 pairs, row=m-in-tile, pair-col=k/2
// sB: [ 64][16] uint = bf16 pairs, row=n-in-tile, pair-col=k/2
// Layouts used here (A/B: contiguous k = 16h + s, not the ISA table order):
//   A 16x32: lane<16 -> M=lane, VGPR j: K=2j,2j+1; lane>=16 -> M=lane-16, K=16+2j
//   B 32x16: lane<16 -> N=lane, VGPR j: K=2j,2j+1; lane>=16 -> N=lane-16, K=16+2j
//   C 16x16: N=lane&15, VGPR j -> M = j + 8*(lane>=16)
__device__ __forceinline__ void wave_mma_step(const unsigned int (*sA)[16],
                                              const unsigned int (*sB)[16],
                                              int mbase, int lane, v8f c[4]) {
  const int hi = lane >> 4;
  const int lc = lane & 15;
  v8u ar;
#pragma unroll
  for (int j = 0; j < 8; ++j) {
    ar[j] = sA[mbase + lc][hi * 8 + j];
  }
#pragma unroll
  for (int a = 0; a < 4; ++a) {
    v8u br;
#pragma unroll
    for (int j = 0; j < 8; ++j) br[j] = sB[a * 16 + lc][hi * 8 + j];
    c[a] = wmma_bf16(ar, br, c[a]);
  }
}

// ---------------------------------------------------------------------------
// K1: per-(b,l) sigma params.  sig = 3^(sigmoid(5*(q.Wsig+b))+1e-5)-1
//     store coef=1/(2*pi*s1*s2), inv1=1/(2*s1^2), inv2=1/(2*s2^2) at [b,h,l]
// ---------------------------------------------------------------------------
__global__ __launch_bounds__(256) void sigma_kernel(
    const float* __restrict__ Q, const float* __restrict__ Wsig,
    const float* __restrict__ bsig, float* __restrict__ coef,
    float* __restrict__ inv1, float* __restrict__ inv2) {
  const int tid  = threadIdx.x;
  const int wave = tid >> 5, lane = tid & 31;
  const int half = lane >> 4, j = lane & 15;          // j = 2h+c
  const int gid  = blockIdx.x * 16 + wave * 2 + half; // row in [0, B*L)
  const float* qrow = Q + (size_t)gid * D_MODEL;
  const float* wrow = Wsig + (size_t)j * D_MODEL;
  float acc = 0.f;
  for (int k = 0; k < D_MODEL; k += 4)
    acc += qrow[k] * wrow[k] + qrow[k + 1] * wrow[k + 1] +
           qrow[k + 2] * wrow[k + 2] + qrow[k + 3] * wrow[k + 3];
  acc += bsig[j];
  const float s  = 1.0f / (1.0f + __expf(-5.0f * acc)) + 1e-5f;
  const float sv = exp2f(s * 1.5849625007211562f) - 1.0f; // 3^s - 1
  const float other = __shfl_xor(sv, 1, 32);
  if ((j & 1) == 0) {
    const float s1 = sv, s2 = other;
    const int h = j >> 1;
    const int b = gid >> 10, l = gid & (L_SZ - 1);
    const size_t idx = ((size_t)(b * H_SZ + h)) * L_SZ + l;
    coef[idx] = 1.0f / (6.283185307179586f * s1 * s2);
    inv1[idx] = 1.0f / (2.0f * s1 * s1);
    inv2[idx] = 1.0f / (2.0f * s2 * s2);
  }
}

// ---------------------------------------------------------------------------
// K2: O_bf16[m,n] = bf16( X[m,:] . W[n,:] + bias[n] )   (M=8192,N=512,K=512)
// ---------------------------------------------------------------------------
__global__ __launch_bounds__(256) void proj_gemm_kernel(
    const float* __restrict__ X, const float* __restrict__ W,
    const float* __restrict__ bias, unsigned short* __restrict__ O) {
  __shared__ unsigned int sA[128][16];
  __shared__ unsigned int sB[64][16];
  const int tid = threadIdx.x;
  const int m0 = blockIdx.x * 128;
  const int n0 = blockIdx.y * 64;
  const int wave = tid >> 5, lane = tid & 31;
  const int hi = lane >> 4, lc = lane & 15;
  v8f zero = {};
  v8f c[4] = {zero, zero, zero, zero};

  const int arow = tid >> 1, ahalf = tid & 1;
  const int brow = tid >> 2, bq = tid & 3;
  const float* aptr = X + (size_t)(m0 + arow) * D_MODEL + ahalf * 16;
  const float* bptr = W + (size_t)(n0 + brow) * D_MODEL + bq * 8;

  for (int kk = 0; kk < D_MODEL; kk += 32) {
    {
      const float4* src = reinterpret_cast<const float4*>(aptr + kk);
#pragma unroll
      for (int p = 0; p < 4; ++p) {
        const float4 f = src[p];
        sA[arow][ahalf * 8 + p * 2 + 0] = f2bf_pack(f.x, f.y);
        sA[arow][ahalf * 8 + p * 2 + 1] = f2bf_pack(f.z, f.w);
      }
    }
    {
      const float4* src = reinterpret_cast<const float4*>(bptr + kk);
#pragma unroll
      for (int p = 0; p < 2; ++p) {
        const float4 f = src[p];
        sB[brow][bq * 4 + p * 2 + 0] = f2bf_pack(f.x, f.y);
        sB[brow][bq * 4 + p * 2 + 1] = f2bf_pack(f.z, f.w);
      }
    }
    if (kk + 32 < D_MODEL) {           // hint next K-step into cache
      __builtin_prefetch(aptr + kk + 32, 0, 1);
      __builtin_prefetch(bptr + kk + 32, 0, 1);
    }
    __syncthreads();
    wave_mma_step(sA, sB, wave * 16, lane, c);
    __syncthreads();
  }
#pragma unroll
  for (int a = 0; a < 4; ++a) {
    const int n = n0 + a * 16 + lc;
    const float bv = bias[n];
#pragma unroll
    for (int j = 0; j < 8; ++j) {
      const int m = m0 + wave * 16 + hi * 8 + j;
      O[(size_t)m * D_MODEL + n] = f2bf(c[a][j] + bv);
    }
  }
}

// ---------------------------------------------------------------------------
// K3a: raw scores  P[bh, l, s] = 0.125 * Q_h[l,:] . K_h[s,:]   (K=64 per head)
// Tiles are already bf16 -> stage them with async global->LDS DMA
// (global_load_async_to_lds_b128, tracked by ASYNCcnt), no VGPR round-trip.
// ---------------------------------------------------------------------------
__global__ __launch_bounds__(256) void attn_scores_kernel(
    const unsigned short* __restrict__ Qbf,
    const unsigned short* __restrict__ Kbf, float* __restrict__ P) {
  __shared__ unsigned int sA[128][16];
  __shared__ unsigned int sB[64][16];
  const int tid = threadIdx.x;
  const int bz = blockIdx.z, b = bz >> 3, h = bz & 7;
  const int m0 = blockIdx.x * 128;
  const int s0 = blockIdx.y * 64;
  const int wave = tid >> 5, lane = tid & 31;
  const int hi = lane >> 4, lc = lane & 15;
  v8f zero = {};
  v8f c[4] = {zero, zero, zero, zero};

  const int arow = tid >> 1, ahalf = tid & 1;   // 2 threads/row, 32B each
  const int brow = tid >> 2, bq = tid & 3;      // 4 threads/row, 16B each

#pragma unroll
  for (int kk = 0; kk < D_HEAD; kk += 32) {
    {
      const unsigned short* g = Qbf +
          (size_t)(b * L_SZ + m0 + arow) * D_MODEL + h * D_HEAD + kk + ahalf * 16;
      const unsigned la = lds_off(&sA[arow][ahalf * 8]);
      // copies 16B to LDS[la], then 16B to LDS[la+16] (offset adds to both sides)
      asm volatile("global_load_async_to_lds_b128 %0, %1, off"
                   :: "v"(la), "v"(g) : "memory");
      asm volatile("global_load_async_to_lds_b128 %0, %1, off offset:16"
                   :: "v"(la), "v"(g) : "memory");
    }
    {
      const unsigned short* g = Kbf +
          (size_t)(b * L_SZ + s0 + brow) * D_MODEL + h * D_HEAD + kk + bq * 8;
      const unsigned lb = lds_off(&sB[brow][bq * 4]);
      asm volatile("global_load_async_to_lds_b128 %0, %1, off"
                   :: "v"(lb), "v"(g) : "memory");
    }
    asm volatile("s_wait_asynccnt 0" ::: "memory");
    __syncthreads();
    wave_mma_step(sA, sB, wave * 16, lane, c);
    __syncthreads();
  }
  const size_t base = (size_t)bz * L_SZ * L_SZ;
#pragma unroll
  for (int a = 0; a < 4; ++a)
#pragma unroll
    for (int j = 0; j < 8; ++j) {
      const int m = m0 + wave * 16 + hi * 8 + j;
      const int s = s0 + a * 16 + lc;
      P[base + (size_t)m * L_SZ + s] = 0.125f * c[a][j];
    }
}

// ---------------------------------------------------------------------------
// K3b: fused softmax (in place over P row) + Gaussian target row
// one 256-thread block per (b,h,l) row of 1024
// ---------------------------------------------------------------------------
__global__ __launch_bounds__(256) void softmax_target_kernel(
    float* __restrict__ P, float* __restrict__ T,
    const float* __restrict__ dy, const float* __restrict__ dx,
    const float* __restrict__ coef, const float* __restrict__ inv1,
    const float* __restrict__ inv2) {
  __shared__ float red[256];
  const int tid = threadIdx.x;
  const int row = blockIdx.x;           // (b*H+h)*L + l
  const int l = row & (L_SZ - 1);
  float* prow = P + (size_t)row * L_SZ;
  float x[4];
  float m = -3.0e38f;
#pragma unroll
  for (int i = 0; i < 4; ++i) { x[i] = prow[tid + i * 256]; m = fmaxf(m, x[i]); }
  red[tid] = m; __syncthreads();
  for (int s = 128; s > 0; s >>= 1) {
    if (tid < s) red[tid] = fmaxf(red[tid], red[tid + s]);
    __syncthreads();
  }
  const float rmax = red[0]; __syncthreads();
  float sum = 0.f;
#pragma unroll
  for (int i = 0; i < 4; ++i) { x[i] = __expf(x[i] - rmax); sum += x[i]; }
  red[tid] = sum; __syncthreads();
  for (int s = 128; s > 0; s >>= 1) {
    if (tid < s) red[tid] += red[tid + s];
    __syncthreads();
  }
  const float inv = 1.0f / red[0];
#pragma unroll
  for (int i = 0; i < 4; ++i) prow[tid + i * 256] = x[i] * inv;

  const float c0 = coef[row], a1 = inv1[row], a2 = inv2[row];
  const float* dyr = dy + (size_t)l * L_SZ;
  const float* dxr = dx + (size_t)l * L_SZ;
  float* trow = T + (size_t)row * L_SZ;
#pragma unroll
  for (int i = 0; i < 4; ++i) {
    const int s = tid + i * 256;
    trow[s] = c0 * __expf(-dyr[s] * a1 - dxr[s] * a2);
  }
}

// ---------------------------------------------------------------------------
// K4: Out[b,l, h*64+dd] = sum_s P[bh,l,s] * V[b,s, h*64+dd]   (M=1024,N=64,K=1024)
// ---------------------------------------------------------------------------
__global__ __launch_bounds__(256) void attn_out_kernel(
    const float* __restrict__ P, const unsigned short* __restrict__ Vbf,
    float* __restrict__ Out) {
  __shared__ unsigned int sA[128][16];
  __shared__ unsigned int sB[64][16];
  const int tid = threadIdx.x;
  const int bz = blockIdx.z, b = bz >> 3, h = bz & 7;
  const int m0 = blockIdx.x * 128;
  const int wave = tid >> 5, lane = tid & 31;
  const int hi = lane >> 4, lc = lane & 15;
  v8f zero = {};
  v8f c[4] = {zero, zero, zero, zero};
  const size_t pbase = (size_t)bz * L_SZ * L_SZ;

  const int arow = tid >> 1, ahalf = tid & 1;
  const float* aptr = P + pbase + (size_t)(m0 + arow) * L_SZ + ahalf * 16;

  for (int kk = 0; kk < L_SZ; kk += 32) {
    {
      const float4* src = reinterpret_cast<const float4*>(aptr + kk);
#pragma unroll
      for (int p = 0; p < 4; ++p) {
        const float4 f = src[p];
        sA[arow][ahalf * 8 + p * 2 + 0] = f2bf_pack(f.x, f.y);
        sA[arow][ahalf * 8 + p * 2 + 1] = f2bf_pack(f.z, f.w);
      }
    }
    {
      // B tile transposed: sB[dd][i] packs V rows (kk+2i, kk+2i+1) at col dd
      const int dd = tid & 63, ib = tid >> 6;
#pragma unroll
      for (int r = 0; r < 4; ++r) {
        const int i = ib * 4 + r;
        const int s = kk + 2 * i;
        const unsigned int v0 =
            Vbf[(size_t)(b * L_SZ + s) * D_MODEL + h * D_HEAD + dd];
        const unsigned int v1 =
            Vbf[(size_t)(b * L_SZ + s + 1) * D_MODEL + h * D_HEAD + dd];
        sB[dd][i] = v0 | (v1 << 16);
      }
    }
    if (kk + 32 < L_SZ) __builtin_prefetch(aptr + kk + 32, 0, 1);
    __syncthreads();
    wave_mma_step(sA, sB, wave * 16, lane, c);
    __syncthreads();
  }
#pragma unroll
  for (int a = 0; a < 4; ++a)
#pragma unroll
    for (int j = 0; j < 8; ++j) {
      const int m = m0 + wave * 16 + hi * 8 + j;
      const int dd = a * 16 + lc;
      Out[(size_t)(b * L_SZ + m) * D_MODEL + h * D_HEAD + dd] = c[a][j];
    }
}

// ---------------------------------------------------------------------------
// Host-side launcher
// inputs (setup_inputs dict order):
//  0 query 1 key 2 value 3 Wq 4 bq 5 Wk 6 bk 7 Wv 8 bv 9 Wsig 10 bsig
//  11 distances_y 12 distances_x
// outputs: out | p | target (fp32, concatenated)
// ---------------------------------------------------------------------------
extern "C" void kernel_launch(void* const* d_in, const int* in_sizes, int n_in,
                              void* d_out, int out_size, void* d_ws,
                              size_t ws_size, hipStream_t stream) {
  (void)in_sizes; (void)n_in; (void)out_size; (void)ws_size;
  const float* q    = (const float*)d_in[0];
  const float* k    = (const float*)d_in[1];
  const float* v    = (const float*)d_in[2];
  const float* Wq   = (const float*)d_in[3];
  const float* bq   = (const float*)d_in[4];
  const float* Wk   = (const float*)d_in[5];
  const float* bk   = (const float*)d_in[6];
  const float* Wv   = (const float*)d_in[7];
  const float* bv   = (const float*)d_in[8];
  const float* Wsig = (const float*)d_in[9];
  const float* bsig = (const float*)d_in[10];
  const float* dy   = (const float*)d_in[11];
  const float* dx   = (const float*)d_in[12];

  float* out = (float*)d_out;                         // [8,1024,512]
  float* P   = out + (size_t)B_SZ * L_SZ * D_MODEL;   // [64,1024,1024]
  float* T   = P + (size_t)B_SZ * H_SZ * L_SZ * L_SZ; // [64,1024,1024]

  const size_t n_rows = (size_t)B_SZ * L_SZ;          // 8192
  unsigned short* Qbf = (unsigned short*)d_ws;
  unsigned short* Kbf = Qbf + n_rows * D_MODEL;
  unsigned short* Vbf = Kbf + n_rows * D_MODEL;
  float* coef = (float*)(Vbf + n_rows * D_MODEL);
  float* inv1 = coef + (size_t)B_SZ * H_SZ * L_SZ;
  float* inv2 = inv1 + (size_t)B_SZ * H_SZ * L_SZ;

  // K1: sigma params (16 rows per 256-thread block)
  sigma_kernel<<<(B_SZ * L_SZ) / 16, 256, 0, stream>>>(q, Wsig, bsig, coef,
                                                       inv1, inv2);
  // K2: Q/K/V projections (bf16 WMMA, fp32 accumulate, bf16 out)
  dim3 gproj(64, 8, 1);
  proj_gemm_kernel<<<gproj, 256, 0, stream>>>(q, Wq, bq, Qbf);
  proj_gemm_kernel<<<gproj, 256, 0, stream>>>(k, Wk, bk, Kbf);
  proj_gemm_kernel<<<gproj, 256, 0, stream>>>(v, Wv, bv, Vbf);
  // K3a: attention scores (async global->LDS staging + WMMA)
  attn_scores_kernel<<<dim3(8, 16, 64), 256, 0, stream>>>(Qbf, Kbf, P);
  // K3b: fused softmax + Gaussian target
  softmax_target_kernel<<<B_SZ * H_SZ * L_SZ, 256, 0, stream>>>(P, T, dy, dx,
                                                                coef, inv1,
                                                                inv2);
  // K4: out = P . V
  attn_out_kernel<<<dim3(8, 1, 64), 256, 0, stream>>>(P, Vbf, out);
}
